// AttentionHead_43095701848098
// MI455X (gfx1250) — matrix operand order repaired and checked by reading, not compiled, not run on hardware
//
#include <hip/hip_runtime.h>
#include <hip/hip_bf16.h>

typedef __attribute__((ext_vector_type(16))) _Float16 v16h;
typedef __attribute__((ext_vector_type(8)))  _Float16 v8h;
typedef __attribute__((ext_vector_type(8)))  float    v8f;
typedef __attribute__((ext_vector_type(4)))  float    v4f;

#define T_SEQ 256
#define HS    64
#define CE    384
#define KV_PITCH 72     // halves per K/V row (64 + pad)
#define QT_PITCH 264    // halves per Qt row (256 + pad)
#define P_PITCH  40     // halves per P-scratch row (32 + pad)

#define SMEM_HALVES (2 * T_SEQ * KV_PITCH + HS * QT_PITCH + 8 * 16 * P_PITCH)
#define SMEM_BYTES  (SMEM_HALVES * 2)   // 117,760 B < 160 KB -> 2 blocks/WGP

__device__ __forceinline__ v8f vzero8() {
  v8f z = {0.f, 0.f, 0.f, 0.f, 0.f, 0.f, 0.f, 0.f};
  return z;
}

__device__ __forceinline__ v16h cat8(v8h lo, v8h hiv) {
  v16h r;
#pragma unroll
  for (int i = 0; i < 8; ++i) { r[i] = lo[i]; r[8 + i] = hiv[i]; }
  return r;
}

// A fragment (16x32 f16) from a row-major f16 source. Per ISA layout:
// lanes 0-15 (M=lane): halves 0-7 = K0-7, halves 8-15 = K16-23;
// lanes 16-31:         halves 0-7 = K8-15, halves 8-15 = K24-31.
__device__ __forceinline__ v16h load_a_f16(const _Float16* base, int pitch,
                                           int k0, int ln, int hi) {
  const _Float16* r = base + ln * pitch + k0;
  v8h g0 = *(const v8h*)(r + (hi ? 8 : 0));
  v8h g1 = *(const v8h*)(r + (hi ? 24 : 16));
  return cat8(g0, g1);
}

// A fragment built from row-major f32 source (inp), converted to f16.
__device__ __forceinline__ v16h load_a_f32cvt(const float* rowp, int k0, int hi) {
  const float* g0 = rowp + k0 + (hi ? 8 : 0);
  const float* g1 = rowp + k0 + (hi ? 24 : 16);
  v4f f0 = *(const v4f*)g0;
  v4f f1 = *(const v4f*)(g0 + 4);
  v4f f2 = *(const v4f*)g1;
  v4f f3 = *(const v4f*)(g1 + 4);
  v16h av;
#pragma unroll
  for (int i = 0; i < 4; ++i) {
    av[i]      = (_Float16)f0[i];
    av[4 + i]  = (_Float16)f1[i];
    av[8 + i]  = (_Float16)f2[i];
    av[12 + i] = (_Float16)f3[i];
  }
  return av;
}

// B fragment (32x16 f16): lane holds column N=lane%16; halves in table K order.
__device__ __forceinline__ v16h load_b_f16(const _Float16* p, int pitch, int hi) {
  v16h r;
#pragma unroll
  for (int s = 0; s < 8; ++s) {
    r[s]     = p[(8 * hi + s) * pitch];
    r[8 + s] = p[(16 + 8 * hi + s) * pitch];
  }
  return r;
}

__device__ __forceinline__ v8f wmma_f16(v16h a, v16h b, v8f c) {
  return __builtin_amdgcn_wmma_f32_16x16x32_f16(false, a, false, b, (short)0, c,
                                                false, false);
}

__device__ __forceinline__ float rmax16(float v) {
  v = fmaxf(v, __shfl_xor(v, 1));
  v = fmaxf(v, __shfl_xor(v, 2));
  v = fmaxf(v, __shfl_xor(v, 4));
  v = fmaxf(v, __shfl_xor(v, 8));
  return v;
}
__device__ __forceinline__ float rsum16(float v) {
  v += __shfl_xor(v, 1);
  v += __shfl_xor(v, 2);
  v += __shfl_xor(v, 4);
  v += __shfl_xor(v, 8);
  return v;
}

__global__ void cvt_weights_kernel(const float* __restrict__ Wv,
                                   const float* __restrict__ Wk,
                                   const float* __restrict__ Wq,
                                   _Float16* __restrict__ Wh) {
  const int i = blockIdx.x * blockDim.x + threadIdx.x;
  const int N = CE * HS;
  if (i < N) {
    Wh[i]         = (_Float16)Wv[i];
    Wh[N + i]     = (_Float16)Wk[i];
    Wh[2 * N + i] = (_Float16)Wq[i];
  }
}

extern __shared__ _Float16 smem[];

__global__ __launch_bounds__(256)
void attn_head_kernel(const float* __restrict__ inp,
                      const _Float16* __restrict__ Wh,
                      const float* __restrict__ bv,
                      const float* __restrict__ bk,
                      const float* __restrict__ bq,
                      float* __restrict__ out) {
  _Float16* Ks = smem;                       // [256][KV_PITCH] row-major keys
  _Float16* Vs = Ks + T_SEQ * KV_PITCH;      // [256][KV_PITCH] row-major values
  _Float16* Qt = Vs + T_SEQ * KV_PITCH;      // [64][QT_PITCH]  queries TRANSPOSED
  _Float16* Pb = Qt + HS * QT_PITCH;         // 8 x [16][P_PITCH] per-wave P scratch

  const int b    = blockIdx.x;
  const int tid  = (int)threadIdx.x;
  const int lane = tid & 31;
  const int w    = tid >> 5;
  const int hi   = lane >> 4;
  const int ln   = lane & 15;
  _Float16* Pw = Pb + w * 16 * P_PITCH;

  const float* inB = inp + (size_t)b * T_SEQ * CE;

  // ---------------- projections: V,K,Q = inp @ W + bias ----------------
  // Software-pipelined K-loop: next iteration's A + 4 B fragments are loaded
  // before the current iteration's 4 WMMAs. The pipelined loop is unrolled by
  // 2 so the rotation is renamed (ping-pong) instead of copied.
#pragma unroll 1
  for (int rep = 0; rep < 2; ++rep) {
    const int tt = rep ? (15 - w) : w;      // pair w with 15-w for balance
    const int t0 = tt * 16;
    const float* rowp = inB + (t0 + ln) * CE;

#pragma unroll 1
    for (int mat = 0; mat < 3; ++mat) {
      const _Float16* Wm = Wh + mat * (CE * HS);
      v8f c[4] = {vzero8(), vzero8(), vzero8(), vzero8()};

      // prologue: fragments for ks = 0
      v16h a  = load_a_f32cvt(rowp, 0, hi);
      const _Float16* wp0 = Wm + ln;
      v16h b0 = load_b_f16(wp0 + 0, HS, hi);
      v16h b1 = load_b_f16(wp0 + 16, HS, hi);
      v16h b2 = load_b_f16(wp0 + 32, HS, hi);
      v16h b3 = load_b_f16(wp0 + 48, HS, hi);

#pragma unroll 2
      for (int ks = 0; ks < 11; ++ks) {
        const int k1 = (ks + 1) * 32;
        __builtin_prefetch(rowp + k1 + 32, 0, 3);
        // issue next-iteration loads first ...
        v16h an = load_a_f32cvt(rowp, k1, hi);
        const _Float16* wpn = Wm + k1 * HS + ln;
        v16h n0 = load_b_f16(wpn + 0, HS, hi);
        v16h n1 = load_b_f16(wpn + 16, HS, hi);
        v16h n2 = load_b_f16(wpn + 32, HS, hi);
        v16h n3 = load_b_f16(wpn + 48, HS, hi);
        // ... then do this iteration's math while they are in flight
        c[0] = wmma_f16(a, b0, c[0]);
        c[1] = wmma_f16(a, b1, c[1]);
        c[2] = wmma_f16(a, b2, c[2]);
        c[3] = wmma_f16(a, b3, c[3]);
        a = an; b0 = n0; b1 = n1; b2 = n2; b3 = n3;
      }
      // epilogue K-step
      c[0] = wmma_f16(a, b0, c[0]);
      c[1] = wmma_f16(a, b1, c[1]);
      c[2] = wmma_f16(a, b2, c[2]);
      c[3] = wmma_f16(a, b3, c[3]);

      const float* bias = (mat == 0) ? bv : ((mat == 1) ? bk : bq);
#pragma unroll
      for (int ct = 0; ct < 4; ++ct) {
        const int n0 = ct * 16;
        const float bb = bias[n0 + ln];
        if (mat == 2) {  // Q -> transposed store: contiguous along M
          v8h q;
#pragma unroll
          for (int j = 0; j < 8; ++j) q[j] = (_Float16)(c[ct][j] + bb);
          *(v8h*)(Qt + (n0 + ln) * QT_PITCH + t0 + 8 * hi) = q;
        } else {         // V / K -> row-major (strided b16 stores)
          _Float16* dst = (mat == 0) ? Vs : Ks;
#pragma unroll
          for (int j = 0; j < 8; ++j)
            dst[(t0 + j + 8 * hi) * KV_PITCH + n0 + ln] =
                (_Float16)(c[ct][j] + bb);
        }
      }
    }
  }
  __syncthreads();

  // ---------------- attention: softmax(mask(K Q^T * 384^-0.5)) V -------------
  const float scale = 0.051031036307982884f;  // 384^-0.5 (embed dim, per ref)
#pragma unroll 1
  for (int rep = 0; rep < 2; ++rep) {
    const int tt = rep ? (15 - w) : w;
    const int t0 = tt * 16;
    const int sMax = t0 + 15;

    v8f o0 = vzero8(), o1 = vzero8(), o2 = vzero8(), o3 = vzero8();
    float m8[8], l8[8];
#pragma unroll
    for (int j = 0; j < 8; ++j) { m8[j] = -3.0e38f; l8[j] = 0.f; }

    // K-tile A fragments are invariant over the s loop: load once.
    const v16h ak0 = load_a_f16(Ks + t0 * KV_PITCH, KV_PITCH, 0, ln, hi);
    const v16h ak1 = load_a_f16(Ks + t0 * KV_PITCH, KV_PITCH, 32, ln, hi);

    for (int s0 = 0; s0 <= sMax; s0 += 32) {
      const bool have1 = (s0 + 16) <= sMax;

      // S0 = K[t0..t0+15] . Q[s0..s0+15]^T  (K dim = 64, two WMMA steps)
      v8f S0 = vzero8();
      S0 = wmma_f16(ak0, load_b_f16(Qt + s0 + ln, QT_PITCH, hi), S0);
      S0 = wmma_f16(ak1, load_b_f16(Qt + 32 * QT_PITCH + s0 + ln, QT_PITCH, hi), S0);
      v8f S1 = vzero8();
      if (have1) {
        S1 = wmma_f16(ak0, load_b_f16(Qt + s0 + 16 + ln, QT_PITCH, hi), S1);
        S1 = wmma_f16(ak1, load_b_f16(Qt + 32 * QT_PITCH + s0 + 16 + ln, QT_PITCH, hi), S1);
      }

      // V B-fragments depend only on s0: issue their LDS loads now so they
      // overlap the softmax VALU work below.
      int vrow = s0;                              // clamp: padded cols of P are 0
      if (vrow > T_SEQ - 1) vrow = T_SEQ - 1;
      const _Float16* vr = Vs + vrow * KV_PITCH + ln;
      v16h vb0 = load_b_f16(vr + 0, KV_PITCH, hi);
      v16h vb1 = load_b_f16(vr + 16, KV_PITCH, hi);
      v16h vb2 = load_b_f16(vr + 32, KV_PITCH, hi);
      v16h vb3 = load_b_f16(vr + 48, KV_PITCH, hi);

      // scale + causal mask + online softmax update + pack P into LDS
#pragma unroll
      for (int j = 0; j < 8; ++j) {
        const int M = j + 8 * hi;                 // row t - t0
        float x0 = S0[j] * scale;
        if (s0 + ln > t0 + M) x0 = -3.0e38f;      // s > t -> masked
        float x1 = -3.0e38f;
        if (have1) {
          x1 = S1[j] * scale;
          if (s0 + 16 + ln > t0 + M) x1 = -3.0e38f;
        }
        float mx   = rmax16(fmaxf(x0, x1));       // row-max across 16 lanes
        float mnew = fmaxf(m8[j], mx);
        float fac  = __expf(m8[j] - mnew);
        float p0   = __expf(x0 - mnew);
        float p1   = have1 ? __expf(x1 - mnew) : 0.f;
        l8[j] = l8[j] * fac + rsum16(p0 + p1);
        m8[j] = mnew;
        o0[j] *= fac; o1[j] *= fac; o2[j] *= fac; o3[j] *= fac;
        Pw[M * P_PITCH + ln]      = (_Float16)p0;
        Pw[M * P_PITCH + 16 + ln] = (_Float16)p1;
      }

      // O += P[16x32] . V[s0..s0+31][0..63]
      v16h ap = load_a_f16(Pw, P_PITCH, 0, ln, hi);
      o0 = wmma_f16(ap, vb0, o0);
      o1 = wmma_f16(ap, vb1, o1);
      o2 = wmma_f16(ap, vb2, o2);
      o3 = wmma_f16(ap, vb3, o3);
    }

    // normalize by row sums and write out (f32)
    float* op = out + ((size_t)b * T_SEQ + t0) * HS;
#pragma unroll
    for (int j = 0; j < 8; ++j) {
      const int M = j + 8 * hi;
      const float inv = 1.0f / l8[j];
      op[M * HS + 0  + ln] = o0[j] * inv;
      op[M * HS + 16 + ln] = o1[j] * inv;
      op[M * HS + 32 + ln] = o2[j] * inv;
      op[M * HS + 48 + ln] = o3[j] * inv;
    }
  }
}

extern "C" void kernel_launch(void* const* d_in, const int* in_sizes, int n_in,
                              void* d_out, int out_size, void* d_ws, size_t ws_size,
                              hipStream_t stream) {
  (void)in_sizes; (void)n_in; (void)out_size; (void)ws_size;
  const float* inp = (const float*)d_in[0];
  const float* Wv  = (const float*)d_in[1];
  const float* bvp = (const float*)d_in[2];
  const float* Wk  = (const float*)d_in[3];
  const float* bkp = (const float*)d_in[4];
  const float* Wq  = (const float*)d_in[5];
  const float* bqp = (const float*)d_in[6];
  _Float16* Wh = (_Float16*)d_ws;          // 3 * 384*64 f16 = 147,456 B

  const int cvtN = CE * HS;
  cvt_weights_kernel<<<(cvtN + 255) / 256, 256, 0, stream>>>(Wv, Wk, Wq, Wh);

  hipFuncSetAttribute(reinterpret_cast<const void*>(attn_head_kernel),
                      hipFuncAttributeMaxDynamicSharedMemorySize, SMEM_BYTES);
  attn_head_kernel<<<512, 256, SMEM_BYTES, stream>>>(inp, Wh, bvp, bkp, bqp,
                                                     (float*)d_out);
}
